// DXVAE_64699387347243
// MI455X (gfx1250) — matrix operand order repaired and checked by reading, not compiled, not run on hardware
//
#include <hip/hip_runtime.h>
#include <hip/hip_bf16.h>
#include <math.h>

// ---------------------------------------------------------------------------
// DXVAE graph-GRU encoder for gfx1250 (MI455X).
// Heavy matmuls via v_wmma_f32_16x16x32_bf16 (f32 accumulate), M=32 tiles per
// wave (2 A-fragments share every B-fragment). The adjacency 0/1 masks are
// factored out of the GEMM: cat@W^T = pred*(h@W_lo^T) + succ*(h@W_hi^T),
// applied per output row in the epilogue -> branch-free inner loops.
// Hidden states carried in f32 with a bf16 shadow used as WMMA A-operands.
// ---------------------------------------------------------------------------

#define B_   8192
#define NN   7
#define SX   27
#define SX0  23
#define SH   512
#define H3   1536
#define SZ   128

typedef __bf16 bf16_t;
typedef __attribute__((ext_vector_type(16))) __bf16 v16bf;
typedef __attribute__((ext_vector_type(8)))  __bf16 v8bf;
typedef __attribute__((ext_vector_type(8)))  float  v8f;

static __device__ __forceinline__ v8f wmma_bf(v16bf a, v16bf b, v8f c) {
  // (neg_a, A, neg_b, B, c_mod, C, reuse_a, reuse_b)
  return __builtin_amdgcn_wmma_f32_16x16x32_bf16(false, a, false, b, (short)0, c,
                                                 false, false);
}

// A fragment (16x32 bf16, M=row,K): lane group g=lane>>4 holds
// K = {g*8 .. g*8+7} and {16+g*8 .. 16+g*8+7} of row (lane&15). (ISA 7.12.2)
static __device__ __forceinline__ v16bf frag_a(const bf16_t* __restrict__ row_kb,
                                               int grp) {
  v8bf c0 = *(const v8bf*)(row_kb + grp * 8);
  v8bf c1 = *(const v8bf*)(row_kb + 16 + grp * 8);
  v16bf r;
#pragma unroll
  for (int i = 0; i < 8; ++i) { r[i] = c0[i]; r[i + 8] = c1[i]; }
  return r;
}

// B fragment (32x16 bf16, K,N=col): lane group g holds
// K = {g*8 .. g*8+7} and {16+g*8 .. 16+g*8+7} of weight row (lane&15) (row-major W[out][in]).
static __device__ __forceinline__ v16bf frag_b(const bf16_t* __restrict__ row_kb,
                                               int grp) {
  v8bf c0 = *(const v8bf*)(row_kb + grp * 8);
  v8bf c1 = *(const v8bf*)(row_kb + 16 + grp * 8);
  v16bf r;
#pragma unroll
  for (int i = 0; i < 8; ++i) { r[i] = c0[i]; r[i + 8] = c1[i]; }
  return r;
}

static __device__ __forceinline__ float sigmoid_(float x) {
  return 1.0f / (1.0f + __expf(-x));
}

// --------------------------- weight conversion -----------------------------
__global__ void __launch_bounds__(256) dxvae_cvt(const float* __restrict__ src,
                                                 bf16_t* __restrict__ dst,
                                                 int rows, int scols, int dcols) {
  int total = rows * dcols;
  for (int i = blockIdx.x * blockDim.x + threadIdx.x; i < total;
       i += gridDim.x * blockDim.x) {
    int r = i / dcols, c = i - r * dcols;
    dst[i] = (c < scols) ? (bf16_t)src[(size_t)r * scols + c] : (bf16_t)0.0f;
  }
}

__global__ void __launch_bounds__(256) dxvae_zero(float* __restrict__ hf,
                                                  bf16_t* __restrict__ hb,
                                                  size_t n) {
  for (size_t i = blockIdx.x * (size_t)blockDim.x + threadIdx.x; i < n;
       i += (size_t)gridDim.x * blockDim.x) {
    hf[i] = 0.0f;
    hb[i] = (bf16_t)0.0f;
  }
}

// --------------------------- gate/map aggregation --------------------------
// H_in[b,:] = sum_{u>v} sigmoid(cat_u @ Wg^T + bg) * (cat_u @ Wm^T)
// cat_u = [h_u*pred, h_u*succ], pred/succ in {0,1}; masks applied in epilogue:
//   cat@W^T = pred*(h@W[:, :512]^T) + succ*(h@W[:, 512:]^T)
__global__ void __launch_bounds__(256) dxvae_gate(
    const bf16_t* __restrict__ hidden_bf,  // [B][7][512]
    const int*    __restrict__ adj,        // [B][7][7]
    const bf16_t* __restrict__ Wg,         // [512][1024]
    const bf16_t* __restrict__ Wm,         // [512][1024]
    const float*  __restrict__ bgate,      // [512]
    float*        __restrict__ hin_f,      // [B][512]
    bf16_t*       __restrict__ hin_b,      // [B][512]
    int v) {
  const int lane  = threadIdx.x & 31;
  const int wave  = threadIdx.x >> 5;
  const int grp   = lane >> 4;
  const int mrow  = lane & 15;
  const int bbase = blockIdx.x * 32;               // 32 batch rows / wave tile
  const int nbase = blockIdx.y * 128 + wave * 16;
  const int j     = nbase + mrow;                  // output col / weight row
  const int ba0   = bbase + mrow;                  // A rows, tile 0
  const int ba1   = bbase + 16 + mrow;             // A rows, tile 1

  const bf16_t* wg_row = Wg + (size_t)j * 1024;    // lo half: +kb, hi: +512+kb
  const bf16_t* wm_row = Wm + (size_t)j * 1024;
  const float   bg     = bgate[j];

  v8f acch0 = {}, acch1 = {};
  for (int u = v + 1; u < NN; ++u) {
    const bf16_t* hrow0 = hidden_bf + ((size_t)ba0 * NN + u) * SH;
    const bf16_t* hrow1 = hidden_bf + ((size_t)ba1 * NN + u) * SH;

    v8f gl0 = {}, gh0 = {}, ml0 = {}, mh0 = {};
    v8f gl1 = {}, gh1 = {}, ml1 = {}, mh1 = {};
#pragma unroll 2
    for (int kb = 0; kb < SH; kb += 32) {
      const v16bf a0   = frag_a(hrow0 + kb, grp);
      const v16bf a1   = frag_a(hrow1 + kb, grp);
      const v16bf bglo = frag_b(wg_row + kb, grp);
      const v16bf bghi = frag_b(wg_row + SH + kb, grp);
      const v16bf bmlo = frag_b(wm_row + kb, grp);
      const v16bf bmhi = frag_b(wm_row + SH + kb, grp);
      gl0 = wmma_bf(a0, bglo, gl0);  gl1 = wmma_bf(a1, bglo, gl1);
      gh0 = wmma_bf(a0, bghi, gh0);  gh1 = wmma_bf(a1, bghi, gh1);
      ml0 = wmma_bf(a0, bmlo, ml0);  ml1 = wmma_bf(a1, bmlo, ml1);
      mh0 = wmma_bf(a0, bmhi, mh0);  mh1 = wmma_bf(a1, bmhi, mh1);
    }
    // epilogue: apply per-row pred/succ (exact; they are 0/1)
#pragma unroll
    for (int r = 0; r < 8; ++r) {
      {
        const int   bm = bbase + r + grp * 8;
        const float fp = (float)adj[((size_t)bm * NN + u) * NN + v];
        const float fs = (float)adj[((size_t)bm * NN + v) * NN + u];
        const float g  = sigmoid_(fp * gl0[r] + fs * gh0[r] + bg);
        acch0[r] += g * (fp * ml0[r] + fs * mh0[r]);
      }
      {
        const int   bm = bbase + 16 + r + grp * 8;
        const float fp = (float)adj[((size_t)bm * NN + u) * NN + v];
        const float fs = (float)adj[((size_t)bm * NN + v) * NN + u];
        const float g  = sigmoid_(fp * gl1[r] + fs * gh1[r] + bg);
        acch1[r] += g * (fp * ml1[r] + fs * mh1[r]);
      }
    }
  }

#pragma unroll
  for (int r = 0; r < 8; ++r) {
    const int bm0 = bbase + r + grp * 8;       // C/D: row r (+8 hi lane group)
    const int bm1 = bm0 + 16;
    hin_f[(size_t)bm0 * SH + j] = acch0[r];
    hin_b[(size_t)bm0 * SH + j] = (bf16_t)acch0[r];
    hin_f[(size_t)bm1 * SH + j] = acch1[r];
    hin_b[(size_t)bm1 * SH + j] = (bf16_t)acch1[r];
  }
}

// --------------------------------- GRU -------------------------------------
__global__ void __launch_bounds__(256) dxvae_gru(
    const float*  __restrict__ X,         // [B][7][27]
    const int*    __restrict__ adj,       // [B][7][7]
    const bf16_t* __restrict__ Wih,       // [1536][32] (K padded)
    const float*  __restrict__ bih,       // [1536]
    const bf16_t* __restrict__ Whh,       // [1536][512]
    const float*  __restrict__ bhh,       // [1536]
    const float*  __restrict__ hprev_f,   // [B][512]
    const bf16_t* __restrict__ hprev_b,   // [B][512]
    float*        __restrict__ out_f,     // [B][512]
    bf16_t*       __restrict__ out_b,     // strided bf16 out
    int out_b_stride, int v, int xdim, int selfloop) {
  const int lane  = threadIdx.x & 31;
  const int wave  = threadIdx.x >> 5;
  const int grp   = lane >> 4;
  const int mrow  = lane & 15;
  const int bbase = blockIdx.x * 32;
  const int nbase = blockIdx.y * 128 + wave * 16;
  const int j     = nbase + mrow;
  const int ba0   = bbase + mrow;
  const int ba1   = bbase + 16 + mrow;

  // x fragments: K padded 27/23 -> 32, fp32 -> bf16, optional self-loop scale
  const float* xrow0 = X + ((size_t)ba0 * NN + v) * SX;
  const float* xrow1 = X + ((size_t)ba1 * NN + v) * SX;
  float xs0 = 1.0f, xs1 = 1.0f;
  if (selfloop) {
    xs0 = (float)adj[((size_t)ba0 * NN + v) * NN + v];
    xs1 = (float)adj[((size_t)ba1 * NN + v) * NN + v];
  }
  v16bf xf0, xf1;
#pragma unroll
  for (int e = 0; e < 16; ++e) {
    const int k = grp * 8 + (e & 7) + ((e >> 3) << 4);
    xf0[e] = (bf16_t)((k < xdim) ? xrow0[k] * xs0 : 0.0f);
    xf1[e] = (bf16_t)((k < xdim) ? xrow1[k] * xs1 : 0.0f);
  }

  v8f air0 = {}, aiz0 = {}, ain0 = {}, ahr0 = {}, ahz0 = {}, ahn0 = {};
  v8f air1 = {}, aiz1 = {}, ain1 = {}, ahr1 = {}, ahz1 = {}, ahn1 = {};
  {
    const v16bf br = frag_b(Wih + (size_t)j * 32, grp);
    const v16bf bz = frag_b(Wih + (size_t)(j + SH) * 32, grp);
    const v16bf bn = frag_b(Wih + (size_t)(j + 2 * SH) * 32, grp);
    air0 = wmma_bf(xf0, br, air0);  air1 = wmma_bf(xf1, br, air1);
    aiz0 = wmma_bf(xf0, bz, aiz0);  aiz1 = wmma_bf(xf1, bz, aiz1);
    ain0 = wmma_bf(xf0, bn, ain0);  ain1 = wmma_bf(xf1, bn, ain1);
  }

  const bf16_t* hrow0 = hprev_b + (size_t)ba0 * SH;
  const bf16_t* hrow1 = hprev_b + (size_t)ba1 * SH;
  const bf16_t* wr = Whh + (size_t)j * SH;
  const bf16_t* wz = Whh + (size_t)(j + SH) * SH;
  const bf16_t* wn = Whh + (size_t)(j + 2 * SH) * SH;
#pragma unroll 2
  for (int kb = 0; kb < SH; kb += 32) {
    const v16bf a0 = frag_a(hrow0 + kb, grp);
    const v16bf a1 = frag_a(hrow1 + kb, grp);
    const v16bf br = frag_b(wr + kb, grp);
    const v16bf bz = frag_b(wz + kb, grp);
    const v16bf bn = frag_b(wn + kb, grp);
    ahr0 = wmma_bf(a0, br, ahr0);  ahr1 = wmma_bf(a1, br, ahr1);
    ahz0 = wmma_bf(a0, bz, ahz0);  ahz1 = wmma_bf(a1, bz, ahz1);
    ahn0 = wmma_bf(a0, bn, ahn0);  ahn1 = wmma_bf(a1, bn, ahn1);
  }

  const float bir = bih[j], biz = bih[j + SH], bin = bih[j + 2 * SH];
  const float bhr = bhh[j], bhz = bhh[j + SH], bhn = bhh[j + 2 * SH];
#pragma unroll
  for (int r = 0; r < 8; ++r) {
    {
      const int   bm = bbase + r + grp * 8;
      const float rg = sigmoid_(air0[r] + bir + ahr0[r] + bhr);
      const float zg = sigmoid_(aiz0[r] + biz + ahz0[r] + bhz);
      const float ng = tanhf(ain0[r] + bin + rg * (ahn0[r] + bhn));
      const float hv = (1.0f - zg) * ng + zg * hprev_f[(size_t)bm * SH + j];
      out_f[(size_t)bm * SH + j] = hv;
      out_b[(size_t)bm * out_b_stride + j] = (bf16_t)hv;
    }
    {
      const int   bm = bbase + 16 + r + grp * 8;
      const float rg = sigmoid_(air1[r] + bir + ahr1[r] + bhr);
      const float zg = sigmoid_(aiz1[r] + biz + ahz1[r] + bhz);
      const float ng = tanhf(ain1[r] + bin + rg * (ahn1[r] + bhn));
      const float hv = (1.0f - zg) * ng + zg * hprev_f[(size_t)bm * SH + j];
      out_f[(size_t)bm * SH + j] = hv;
      out_b[(size_t)bm * out_b_stride + j] = (bf16_t)hv;
    }
  }
}

// --------------------------------- head ------------------------------------
__global__ void __launch_bounds__(256) dxvae_head(
    const bf16_t* __restrict__ h0, int h0_stride,     // hidden_bf node 0
    const bf16_t* __restrict__ Wmu, const float* __restrict__ bmu,
    const bf16_t* __restrict__ Wstd, const float* __restrict__ bstd,
    float* __restrict__ out) {                        // [B][256] = [mu|softplus]
  const int lane  = threadIdx.x & 31;
  const int wave  = threadIdx.x >> 5;
  const int grp   = lane >> 4;
  const int mrow  = lane & 15;
  const int bbase = blockIdx.x * 16;
  const int nbase = blockIdx.y * 128 + wave * 16;     // 0..255 step 16
  const int j     = nbase + mrow;
  const bool is_std = (nbase >= SZ);
  const int  jj   = is_std ? (j - SZ) : j;
  const bf16_t* wrow = (is_std ? Wstd : Wmu) + (size_t)jj * SH;
  const bf16_t* hrow = h0 + (size_t)(bbase + mrow) * h0_stride;

  v8f acc = {};
#pragma unroll 4
  for (int kb = 0; kb < SH; kb += 32)
    acc = wmma_bf(frag_a(hrow + kb, grp), frag_b(wrow + kb, grp), acc);

  const float bias = is_std ? bstd[jj] : bmu[jj];
#pragma unroll
  for (int r = 0; r < 8; ++r) {
    const int bm = bbase + r + grp * 8;
    float y = acc[r] + bias;
    if (is_std) y = (y > 20.0f) ? y : log1pf(__expf(y));
    out[(size_t)bm * (2 * SZ) + j] = y;
  }
}

// ------------------------------- launcher ----------------------------------
extern "C" void kernel_launch(void* const* d_in, const int* in_sizes, int n_in,
                              void* d_out, int out_size, void* d_ws, size_t ws_size,
                              hipStream_t stream) {
  (void)in_sizes; (void)n_in; (void)out_size; (void)ws_size;
  const float* X    = (const float*)d_in[0];
  const int*   adj  = (const int*)d_in[1];
  const float* bcih = (const float*)d_in[4];
  const float* bchh = (const float*)d_in[5];
  const float* blih = (const float*)d_in[8];
  const float* blhh = (const float*)d_in[9];
  const float* brih = (const float*)d_in[12];
  const float* brhh = (const float*)d_in[13];
  const float* bgat = (const float*)d_in[15];
  const float* bmu  = (const float*)d_in[18];
  const float* bstd = (const float*)d_in[20];

  // workspace carve-out (~117 MB total)
  size_t off = 0;
  auto carve = [&](size_t bytes) -> void* {
    void* p = (char*)d_ws + off;
    off += (bytes + 255) & ~(size_t)255;
    return p;
  };
  bf16_t* hidden_bf = (bf16_t*)carve((size_t)B_ * NN * SH * 2);
  float*  hin_f     = (float*) carve((size_t)B_ * SH * 4);
  bf16_t* hin_b     = (bf16_t*)carve((size_t)B_ * SH * 2);
  float*  htmp_f    = (float*) carve((size_t)B_ * SH * 4);
  bf16_t* htmp_b    = (bf16_t*)carve((size_t)B_ * SH * 2);
  bf16_t* Wg   = (bf16_t*)carve((size_t)SH * 1024 * 2);
  bf16_t* Wm   = (bf16_t*)carve((size_t)SH * 1024 * 2);
  bf16_t* Wchh = (bf16_t*)carve((size_t)H3 * SH * 2);
  bf16_t* Wlhh = (bf16_t*)carve((size_t)H3 * SH * 2);
  bf16_t* Wrhh = (bf16_t*)carve((size_t)H3 * SH * 2);
  bf16_t* Wcih = (bf16_t*)carve((size_t)H3 * 32 * 2);
  bf16_t* Wlih = (bf16_t*)carve((size_t)H3 * 32 * 2);
  bf16_t* Wrih = (bf16_t*)carve((size_t)H3 * 32 * 2);
  bf16_t* Wmu  = (bf16_t*)carve((size_t)SZ * SH * 2);
  bf16_t* Wstd = (bf16_t*)carve((size_t)SZ * SH * 2);

  auto cvt = [&](const void* s, bf16_t* d, int rows, int sc, int dc) {
    int total = rows * dc;
    int blocks = (total + 255) / 256;
    if (blocks > 2048) blocks = 2048;
    dxvae_cvt<<<blocks, 256, 0, stream>>>((const float*)s, d, rows, sc, dc);
  };
  cvt(d_in[14], Wg,   SH, 1024, 1024);
  cvt(d_in[16], Wm,   SH, 1024, 1024);
  cvt(d_in[3],  Wchh, H3, SH, SH);
  cvt(d_in[7],  Wlhh, H3, SH, SH);
  cvt(d_in[11], Wrhh, H3, SH, SH);
  cvt(d_in[2],  Wcih, H3, SX,  32);
  cvt(d_in[6],  Wlih, H3, SX,  32);
  cvt(d_in[10], Wrih, H3, SX0, 32);
  cvt(d_in[17], Wmu,  SZ, SH, SH);
  cvt(d_in[19], Wstd, SZ, SH, SH);

  dxvae_zero<<<2048, 256, 0, stream>>>(hin_f, hin_b, (size_t)B_ * SH);

  const dim3 blk(256);
  const dim3 grd(B_ / 32, SH / 128);   // 8 waves/block, 32x16 tile per wave
  for (int v = NN - 1; v >= 0; --v) {
    if (v < NN - 1)
      dxvae_gate<<<grd, blk, 0, stream>>>(hidden_bf, adj, Wg, Wm, bgat,
                                          hin_f, hin_b, v);
    if (v > 0) {
      // cell GRU: h' = GRU(X[:,v], H_in) -> htmp
      dxvae_gru<<<grd, blk, 0, stream>>>(X, adj, Wcih, bcih, Wchh, bchh,
                                         hin_f, hin_b, htmp_f, htmp_b, SH,
                                         v, SX, 0);
      // loop GRU: hidden[:,v] = GRU(X[:,v]*adj[v,v], htmp)
      dxvae_gru<<<grd, blk, 0, stream>>>(X, adj, Wlih, blih, Wlhh, blhh,
                                         htmp_f, htmp_b, hin_f /*f32 dump*/,
                                         hidden_bf + (size_t)v * SH, NN * SH,
                                         v, SX, 1);
    } else {
      // root GRU: hidden[:,0] = GRU(X[:,0,:23], H_in)
      dxvae_gru<<<grd, blk, 0, stream>>>(X, adj, Wrih, brih, Wrhh, brhh,
                                         hin_f, hin_b, htmp_f /*f32 dump*/,
                                         hidden_bf /*node 0*/, NN * SH,
                                         0, SX0, 0);
    }
  }

  const dim3 grd_head(B_ / 16, (2 * SZ) / 128);
  dxvae_head<<<grd_head, blk, 0, stream>>>(hidden_bf, NN * SH,
                                           Wmu, bmu, Wstd, bstd,
                                           (float*)d_out);
}
